// MEBNETBlock_65077344469529
// MI455X (gfx1250) — matrix operand order repaired and checked by reading, not compiled, not run on hardware
//
#include <hip/hip_runtime.h>
#include <hip/hip_bf16.h>

typedef __bf16 bf16;
typedef __attribute__((ext_vector_type(16))) __bf16 v16bf;
typedef __attribute__((ext_vector_type(8)))  __bf16 v8bf;
typedef __attribute__((ext_vector_type(8)))  float  v8f;
typedef __attribute__((ext_vector_type(4)))  float  v4f;

#define SSP_LOG2 0.6931471805599453f

__device__ __forceinline__ float sspf(float x) {
    // numerically stable softplus(x) - log(2)
    float ax = __builtin_fabsf(x);
    return fmaxf(x, 0.0f) + log1pf(__expf(-ax)) - SSP_LOG2;
}

// ---------------------------------------------------------------------------
// Utility kernels
// ---------------------------------------------------------------------------
__global__ void k_zero_f32(float* p, long long n) {
    long long i = (long long)blockIdx.x * blockDim.x + threadIdx.x;
    if (i < n) p[i] = 0.0f;
}

__global__ void k_cast_f32_bf16(const float* __restrict__ in, bf16* __restrict__ out, long long n) {
    long long i = (long long)blockIdx.x * blockDim.x + threadIdx.x;
    if (i < n) out[i] = (bf16)in[i];
}

__global__ void k_gather_batch(const int* __restrict__ src, const int* __restrict__ batch,
                               int* __restrict__ be, int E) {
    int i = blockIdx.x * blockDim.x + threadIdx.x;
    if (i < E) be[i] = batch[src[i]];
}

__global__ void k_count(const int* __restrict__ idx, float* __restrict__ cnt, int n) {
    int i = blockIdx.x * blockDim.x + threadIdx.x;
    if (i < n) atomicAdd(&cnt[idx[i]], 1.0f);
}

// scatter-add of bf16 values [E x C] into f32 acc [S x C] by per-row index
__global__ void k_scatter_add(const bf16* __restrict__ vals, const int* __restrict__ idx,
                              float* __restrict__ acc, long long E, int C) {
    long long i = (long long)blockIdx.x * blockDim.x + threadIdx.x;
    long long total = E * (long long)C;
    if (i < total) {
        long long e = i / C;
        int c = (int)(i - e * C);
        atomicAdd(&acc[(long long)idx[e] * C + c], (float)vals[i]);
    }
}

__global__ void k_mean_bf16(const float* __restrict__ acc, const float* __restrict__ cnt,
                            bf16* __restrict__ out, long long S, int C) {
    long long i = (long long)blockIdx.x * blockDim.x + threadIdx.x;
    long long total = S * (long long)C;
    if (i < total) {
        long long r = i / C;
        out[i] = (bf16)(acc[i] / fmaxf(cnt[r], 1.0f));
    }
}

// ---------------------------------------------------------------------------
// WMMA GEMM:  D[m, o] = ssp( sum_k A[m,k] * W[o,k] )  (+ optional residual)
//  A-loader MODEs (template parameter -> straight-line staging code):
//   0: A = f32  [M,K]   (converted to bf16 on the fly)
//   1: A = bf16 [M,K]
//   2: edge concat4:  [ A0[g0[m]] | A0[g1[m]] | A1[m] | A2[g2[m]] ]  (each 256 wide)
//   3: node concat3:  [ A0[m]     | A1[m]     | A2[g2[m]] ]
//   4: glob concat3:  [ A0[m]     | A1[m]     | A2[m] ]
//
//  Block tile 64x64, K chunks of 64, bf16 in LDS, double-buffered:
//   - B (weights) staged with global_load_async_to_lds_b128 (ASYNCcnt)
//   - A staged via registers: loads issued before compute, stored after
//  8 waves: wave (w&3) -> M tile, (w>>2) -> 2 N tiles (2 independent accs).
//  A and B tiles both K-contiguous per lane -> ds_load_b128 fragments.
// ---------------------------------------------------------------------------
#define BM 64
#define BN 64
#define KC 64
#define PAD 8

template <int MODE>
__global__ __launch_bounds__(256)
void k_gemm_wmma(const float* __restrict__ Af,
                 const bf16* __restrict__ A0, const bf16* __restrict__ A1, const bf16* __restrict__ A2,
                 const int* __restrict__ g0, const int* __restrict__ g1, const int* __restrict__ g2,
                 const bf16* __restrict__ W,
                 int M, int K, int OUT,
                 float* __restrict__ outF, const float* __restrict__ resid,
                 bf16* __restrict__ outB)
{
    __shared__ alignas(16) bf16 As[2][BM][KC + PAD];   // M-major
    __shared__ alignas(16) bf16 Bt[2][BN][KC + PAD];   // N-major: Bt[n][k] = W[n0+n][k0+k]

    const int tid  = threadIdx.x;
    const int m0   = blockIdx.x * BM;
    const int n0   = blockIdx.y * BN;
    const int wave = tid >> 5;
    const int lane = tid & 31;
    const int hl   = lane >> 4;         // lane half
    const int lr   = lane & 15;
    const int mo   = (wave & 3) << 4;   // wave tile row in block tile
    const int no   = (wave >> 2) << 5;  // wave tile col base (2 tiles of 16)

    v8f acc0 = {0.f, 0.f, 0.f, 0.f, 0.f, 0.f, 0.f, 0.f};
    v8f acc1 = {0.f, 0.f, 0.f, 0.f, 0.f, 0.f, 0.f, 0.f};

    // Per-thread staging slots: 2 x 16B for A, 2 x 16B for B per chunk.
    v8bf areg[2];

    // ---- B tile: async DMA global -> LDS (CDNA5), always in-bounds ----
    auto issueB = [&](int k0, int buf) {
        #pragma unroll
        for (int t = 0; t < 2; ++t) {
            int i  = tid + t * 256;
            int n  = i >> 3;
            int kv = (i & 7) << 3;
            unsigned lds = (unsigned)(unsigned long long)&Bt[buf][n][kv];
            unsigned long long ga =
                (unsigned long long)&W[(long long)(n0 + n) * K + (k0 + kv)];
            asm volatile("global_load_async_to_lds_b128 %0, %1, off"
                         :: "v"(lds), "v"(ga) : "memory");
        }
    };

    // ---- A tile: issue global loads into registers (mode-specialized) ----
    auto loadA = [&](int k0) {
        #pragma unroll
        for (int t = 0; t < 2; ++t) {
            int i   = tid + t * 256;
            int m   = i >> 3;
            int kv  = (i & 7) << 3;
            int row = m0 + m;
            int kg  = k0 + kv;
            v8bf val = {};
            if (row < M) {
                if (MODE == 0) {
                    v4f f0 = *(const v4f*)&Af[(long long)row * K + kg];
                    v4f f1 = *(const v4f*)&Af[(long long)row * K + kg + 4];
                    #pragma unroll
                    for (int j = 0; j < 4; ++j) {
                        val[j]     = (bf16)f0[j];
                        val[4 + j] = (bf16)f1[j];
                    }
                } else if (MODE == 1) {
                    val = *(const v8bf*)&A0[(long long)row * K + kg];
                } else {
                    int seg = kg >> 8, kk = kg & 255;
                    const bf16* sp;
                    if (MODE == 2) {
                        if (seg == 0)      sp = &A0[(long long)g0[row] * 256 + kk];
                        else if (seg == 1) sp = &A0[(long long)g1[row] * 256 + kk];
                        else if (seg == 2) sp = &A1[(long long)row * 256 + kk];
                        else               sp = &A2[(long long)g2[row] * 256 + kk];
                    } else if (MODE == 3) {
                        if (seg == 0)      sp = &A0[(long long)row * 256 + kk];
                        else if (seg == 1) sp = &A1[(long long)row * 256 + kk];
                        else               sp = &A2[(long long)g2[row] * 256 + kk];
                    } else {
                        if (seg == 0)      sp = &A0[(long long)row * 256 + kk];
                        else if (seg == 1) sp = &A1[(long long)row * 256 + kk];
                        else               sp = &A2[(long long)row * 256 + kk];
                    }
                    val = *(const v8bf*)sp;
                }
            }
            areg[t] = val;
        }
    };
    auto storeA = [&](int buf) {
        #pragma unroll
        for (int t = 0; t < 2; ++t) {
            int i  = tid + t * 256;
            int m  = i >> 3;
            int kv = (i & 7) << 3;
            *(v8bf*)&As[buf][m][kv] = areg[t];
        }
    };

    // ---- prologue: stage chunk 0 into buffer 0 ----
    issueB(0, 0);
    loadA(0);
    storeA(0);
    asm volatile("s_wait_asynccnt 0x0" ::: "memory");
    __syncthreads();

    const int nchunks = K / KC;
    for (int c = 0; c < nchunks; ++c) {
        const int cur = c & 1;
        const int nxt = cur ^ 1;
        if (c + 1 < nchunks) {
            issueB((c + 1) * KC, nxt);   // async DMA overlaps compute
            loadA((c + 1) * KC);         // loads in flight during WMMAs
        }

        // ---- 4 WMMAs per chunk on buffer `cur` ----
        #pragma unroll
        for (int kk0 = 0; kk0 < KC; kk0 += 32) {
            // A 16x32 bf16 fragment: a[0..7] = K hl*8.., a[8..15] = K 16+hl*8..
            v8bf alo = *(const v8bf*)&As[cur][mo + lr][kk0 + hl * 8];
            v8bf ahi = *(const v8bf*)&As[cur][mo + lr][kk0 + 16 + hl * 8];
            v16bf a = __builtin_shufflevector(alo, ahi,
                0, 1, 2, 3, 4, 5, 6, 7, 8, 9, 10, 11, 12, 13, 14, 15);
            // B 32x16 bf16 fragment: b[0..7] = K hl*8.., b[8..15] = K 16+hl*8..
            v8bf b0lo = *(const v8bf*)&Bt[cur][no + lr][kk0 + hl * 8];
            v8bf b0hi = *(const v8bf*)&Bt[cur][no + lr][kk0 + 16 + hl * 8];
            v16bf b0 = __builtin_shufflevector(b0lo, b0hi,
                0, 1, 2, 3, 4, 5, 6, 7, 8, 9, 10, 11, 12, 13, 14, 15);
            v8bf b1lo = *(const v8bf*)&Bt[cur][no + 16 + lr][kk0 + hl * 8];
            v8bf b1hi = *(const v8bf*)&Bt[cur][no + 16 + lr][kk0 + 16 + hl * 8];
            v16bf b1 = __builtin_shufflevector(b1lo, b1hi,
                0, 1, 2, 3, 4, 5, 6, 7, 8, 9, 10, 11, 12, 13, 14, 15);

            acc0 = __builtin_amdgcn_wmma_f32_16x16x32_bf16(
                false, a, false, b0, (short)0, acc0, false, false);
            acc1 = __builtin_amdgcn_wmma_f32_16x16x32_bf16(
                false, a, false, b1, (short)0, acc1, false, false);
        }

        if (c + 1 < nchunks) storeA(nxt);
        asm volatile("s_wait_asynccnt 0x0" ::: "memory");
        __syncthreads();
    }

    // ---- epilogue: ssp (+residual); C/D layout: VGPR r -> M = r + 8*half ----
    #pragma unroll
    for (int r = 0; r < 8; ++r) {
        int gm = m0 + mo + hl * 8 + r;
        if (gm < M) {
            int gn0 = n0 + no + lr;
            int gn1 = gn0 + 16;
            long long o0 = (long long)gm * OUT + gn0;
            long long o1 = (long long)gm * OUT + gn1;
            float s0 = sspf(acc0[r]);
            float s1 = sspf(acc1[r]);
            if (outF) {
                outF[o0] = s0 + (resid ? resid[o0] : 0.0f);
                outF[o1] = s1 + (resid ? resid[o1] : 0.0f);
            }
            if (outB) {
                outB[o0] = (bf16)s0;
                outB[o1] = (bf16)s1;
            }
        }
    }
}

// ---------------------------------------------------------------------------
// Host orchestration
// ---------------------------------------------------------------------------
static inline unsigned gblk(long long n) { return (unsigned)((n + 255) / 256); }

extern "C" void kernel_launch(void* const* d_in, const int* in_sizes, int n_in,
                              void* d_out, int out_size, void* d_ws, size_t ws_size,
                              hipStream_t stream) {
    (void)in_sizes; (void)n_in; (void)out_size; (void)ws_size;
    constexpr int N = 20000, E = 200000, G = 128, C = 256;

    const float* node_feats = (const float*)d_in[0];
    const float* edge_feats = (const float*)d_in[1];
    const float* glob_feats = (const float*)d_in[2];
    const int*   edge_index = (const int*)d_in[3];
    const int*   batch      = (const int*)d_in[4];
    const float* W_node1    = (const float*)d_in[5];
    const float* W_edge1    = (const float*)d_in[6];
    const float* W_glob1    = (const float*)d_in[7];
    const float* W_edge_mlp = (const float*)d_in[8];
    const float* W_node_mlp = (const float*)d_in[9];
    const float* W_glob_mlp = (const float*)d_in[10];
    const float* W_node2    = (const float*)d_in[11];
    const float* W_edge2    = (const float*)d_in[12];
    const float* W_glob2    = (const float*)d_in[13];

    const int* idx_src = edge_index;
    const int* idx_dst = edge_index + E;

    // ---- workspace carve-up ----
    char* base = (char*)d_ws;
    size_t off = 0;
    auto carve = [&](size_t bytes) -> void* {
        void* p = base + off;
        off = (off + bytes + 255) & ~(size_t)255;
        return p;
    };
    // bf16 weights
    bf16* w1n = (bf16*)carve((size_t)C * C * 2);
    bf16* w1e = (bf16*)carve((size_t)C * C * 2);
    bf16* w1g = (bf16*)carve((size_t)C * C * 2);
    bf16* wem = (bf16*)carve((size_t)C * 4 * C * 2);
    bf16* wnm = (bf16*)carve((size_t)C * 3 * C * 2);
    bf16* wgm = (bf16*)carve((size_t)C * 3 * C * 2);
    bf16* w2n = (bf16*)carve((size_t)C * C * 2);
    bf16* w2e = (bf16*)carve((size_t)C * C * 2);
    bf16* w2g = (bf16*)carve((size_t)C * C * 2);
    // bf16 activations
    bf16* n1b  = (bf16*)carve((size_t)N * C * 2);
    bf16* e1b  = (bf16*)carve((size_t)E * C * 2);
    bf16* g1b  = (bf16*)carve((size_t)G * C * 2);
    bf16* emb  = (bf16*)carve((size_t)E * C * 2);
    bf16* nmb  = (bf16*)carve((size_t)N * C * 2);
    bf16* gmb  = (bf16*)carve((size_t)G * C * 2);
    bf16* e2nb = (bf16*)carve((size_t)N * C * 2);
    bf16* e2gb = (bf16*)carve((size_t)G * C * 2);
    bf16* n2gb = (bf16*)carve((size_t)G * C * 2);
    // contiguous f32 zero zone: accumulators + counts
    long long nzero = (long long)N * C + (long long)G * C * 2 + N + 2 * G;
    float* zzone   = (float*)carve((size_t)nzero * 4);
    float* e2n_acc = zzone;
    float* e2g_acc = e2n_acc + (long long)N * C;
    float* n2g_acc = e2g_acc + (long long)G * C;
    float* cnt_n   = n2g_acc + (long long)G * C;
    float* cnt_eg  = cnt_n + N;
    float* cnt_ng  = cnt_eg + G;
    int* be = (int*)carve((size_t)E * 4);

    float* out_n = (float*)d_out;
    float* out_e = out_n + (long long)N * C;
    float* out_g = out_e + (long long)E * C;

    // ---- weights -> bf16 ----
    k_cast_f32_bf16<<<gblk((long long)C * C), 256, 0, stream>>>(W_node1, w1n, (long long)C * C);
    k_cast_f32_bf16<<<gblk((long long)C * C), 256, 0, stream>>>(W_edge1, w1e, (long long)C * C);
    k_cast_f32_bf16<<<gblk((long long)C * C), 256, 0, stream>>>(W_glob1, w1g, (long long)C * C);
    k_cast_f32_bf16<<<gblk((long long)C * 4 * C), 256, 0, stream>>>(W_edge_mlp, wem, (long long)C * 4 * C);
    k_cast_f32_bf16<<<gblk((long long)C * 3 * C), 256, 0, stream>>>(W_node_mlp, wnm, (long long)C * 3 * C);
    k_cast_f32_bf16<<<gblk((long long)C * 3 * C), 256, 0, stream>>>(W_glob_mlp, wgm, (long long)C * 3 * C);
    k_cast_f32_bf16<<<gblk((long long)C * C), 256, 0, stream>>>(W_node2, w2n, (long long)C * C);
    k_cast_f32_bf16<<<gblk((long long)C * C), 256, 0, stream>>>(W_edge2, w2e, (long long)C * C);
    k_cast_f32_bf16<<<gblk((long long)C * C), 256, 0, stream>>>(W_glob2, w2g, (long long)C * C);

    // ---- zero accumulators (ws is poisoned), be = batch[src] ----
    k_zero_f32<<<gblk(nzero), 256, 0, stream>>>(zzone, nzero);
    k_gather_batch<<<gblk(E), 256, 0, stream>>>(idx_src, batch, be, E);

    // ---- stage 1: lin1 + ssp  (f32 A converted on load) ----
    dim3 blk(256);
    k_gemm_wmma<0><<<dim3((N + BM - 1) / BM, C / BN), blk, 0, stream>>>(
        node_feats, nullptr, nullptr, nullptr, nullptr, nullptr, nullptr,
        w1n, N, C, C, nullptr, nullptr, n1b);
    k_gemm_wmma<0><<<dim3((E + BM - 1) / BM, C / BN), blk, 0, stream>>>(
        edge_feats, nullptr, nullptr, nullptr, nullptr, nullptr, nullptr,
        w1e, E, C, C, nullptr, nullptr, e1b);
    k_gemm_wmma<0><<<dim3((G + BM - 1) / BM, C / BN), blk, 0, stream>>>(
        glob_feats, nullptr, nullptr, nullptr, nullptr, nullptr, nullptr,
        w1g, G, C, C, nullptr, nullptr, g1b);

    // ---- EdgeModel: fused gather-concat4 GEMM (K = 1024) ----
    k_gemm_wmma<2><<<dim3((E + BM - 1) / BM, C / BN), blk, 0, stream>>>(
        nullptr, n1b, e1b, g1b, idx_src, idx_dst, be,
        wem, E, 4 * C, C, nullptr, nullptr, emb);

    // ---- scatter-means from edges ----
    k_scatter_add<<<gblk((long long)E * C), 256, 0, stream>>>(emb, idx_dst, e2n_acc, E, C);
    k_count<<<gblk(E), 256, 0, stream>>>(idx_dst, cnt_n, E);
    k_scatter_add<<<gblk((long long)E * C), 256, 0, stream>>>(emb, be, e2g_acc, E, C);
    k_count<<<gblk(E), 256, 0, stream>>>(be, cnt_eg, E);
    k_mean_bf16<<<gblk((long long)N * C), 256, 0, stream>>>(e2n_acc, cnt_n, e2nb, N, C);

    // ---- NodeModel: fused concat3 GEMM (K = 768) ----
    k_gemm_wmma<3><<<dim3((N + BM - 1) / BM, C / BN), blk, 0, stream>>>(
        nullptr, n1b, e2nb, g1b, nullptr, nullptr, batch,
        wnm, N, 3 * C, C, nullptr, nullptr, nmb);

    // ---- GlobalModel scatter-means + GEMM ----
    k_scatter_add<<<gblk((long long)N * C), 256, 0, stream>>>(nmb, batch, n2g_acc, N, C);
    k_count<<<gblk(N), 256, 0, stream>>>(batch, cnt_ng, N);
    k_mean_bf16<<<gblk((long long)G * C), 256, 0, stream>>>(e2g_acc, cnt_eg, e2gb, G, C);
    k_mean_bf16<<<gblk((long long)G * C), 256, 0, stream>>>(n2g_acc, cnt_ng, n2gb, G, C);
    k_gemm_wmma<4><<<dim3((G + BM - 1) / BM, C / BN), blk, 0, stream>>>(
        nullptr, n2gb, e2gb, g1b, nullptr, nullptr, nullptr,
        wgm, G, 3 * C, C, nullptr, nullptr, gmb);

    // ---- lin2 + ssp + residual -> d_out ----
    k_gemm_wmma<1><<<dim3((N + BM - 1) / BM, C / BN), blk, 0, stream>>>(
        nullptr, nmb, nullptr, nullptr, nullptr, nullptr, nullptr,
        w2n, N, C, C, out_n, node_feats, nullptr);
    k_gemm_wmma<1><<<dim3((E + BM - 1) / BM, C / BN), blk, 0, stream>>>(
        nullptr, emb, nullptr, nullptr, nullptr, nullptr, nullptr,
        w2e, E, C, C, out_e, edge_feats, nullptr);
    k_gemm_wmma<1><<<dim3((G + BM - 1) / BM, C / BN), blk, 0, stream>>>(
        nullptr, gmb, nullptr, nullptr, nullptr, nullptr, nullptr,
        w2g, G, C, C, out_g, glob_feats, nullptr);
}
